// PolyHashV12_71184787964448
// MI455X (gfx1250) — matrix operand order repaired and checked by reading, not compiled, not run on hardware
//
#include <hip/hip_runtime.h>

// ---------------------------------------------------------------------------
// Types for CDNA5 WMMA (wave32, gfx1250)
// ---------------------------------------------------------------------------
typedef __bf16 bf16;
typedef __attribute__((ext_vector_type(16))) __bf16 v16bf;
typedef __attribute__((ext_vector_type(8)))  __bf16 v8bf;
typedef __attribute__((ext_vector_type(8)))  float  v8f;

#define NEG_INF (-1e30f)

// ---------------------------------------------------------------------------
// Problem constants (match the reference)
// ---------------------------------------------------------------------------
constexpr int Bc     = 8;
constexpr int Tc     = 2048;
constexpr int Dc     = 768;          // NUM_TABLES * EMB == SCAN
constexpr int MT     = Bc * Tc;      // 16384 token rows
constexpr int NTAB   = 8;
constexpr int EMB    = 96;
constexpr int NHEAD  = 12;
constexpr int HD     = 64;
constexpr int PTS    = 64;
constexpr int SUBK   = 512;
constexpr int KEYD   = 128;
constexpr int VALD   = 256;
constexpr int TOPK   = 16;

__constant__ unsigned c_primes[8] = {
    2654435761u, 2246822519u, 3266489917u, 2028178513u,
    1220703125u, 1610612741u,  805306457u,  402653189u };

// ---------------------------------------------------------------------------
// fp32 -> bf16 conversion (weights / small tensors)
// ---------------------------------------------------------------------------
__global__ void cvt_bf16_kernel(const float* __restrict__ in, bf16* __restrict__ out, long n) {
    long i = (long)blockIdx.x * blockDim.x + threadIdx.x;
    long stride = (long)gridDim.x * blockDim.x;
    for (; i < n; i += stride) out[i] = (bf16)in[i];
}

// ---------------------------------------------------------------------------
// Multi-scale hash embedding gather.
// Only low 16 bits of the int64 hash survive (% 65536), and all intermediate
// values are non-negative, so 32-bit modular arithmetic is exact.
// One workgroup per (b,t): lanes 0..7 compute the 8 table hashes, then all
// 256 threads gather the 768-float row (fp32 + bf16 shadow).
// ---------------------------------------------------------------------------
__global__ __launch_bounds__(256) void hash_embed_kernel(
    const int* __restrict__ tokens, const float* __restrict__ tables,
    float* __restrict__ xf, bf16* __restrict__ xh)
{
    int bt = blockIdx.x;
    int b = bt / Tc, t = bt % Tc;
    __shared__ int hidx[NTAB];
    int tid = threadIdx.x;
    if (tid < NTAB) {
        int win = 1 << tid;                      // SCALE_WINDOWS[i] == 2^i
        unsigned h = 0;
        for (int j = 0; j < win; ++j) {
            int off = j + 1;
            if (off > t) continue;               // padded region contributes 0
            unsigned tok = (unsigned)tokens[b * Tc + (t - off)];
            h ^= tok * c_primes[j & 7];
        }
        hidx[tid] = (int)(h & 0xFFFFu);
    }
    __syncthreads();
    for (int i = tid; i < Dc; i += blockDim.x) {
        int tab = i / EMB, e = i % EMB;
        float v = tables[((long)tab * 65536 + hidx[tab]) * EMB + e];
        long o = (long)bt * Dc + i;
        xf[o] = v;
        xh[o] = (bf16)v;
    }
}

// ---------------------------------------------------------------------------
// Generic batched/strided bf16 WMMA GEMM:  C = scale * A x B(^T) + bias + resid
//   - 256 threads = 8 waves, wave grid 4(M) x 2(N); each wave owns 16x32 of C
//   - tile 64x64x32, v_wmma_f32_16x16x32_bf16, f32 accumulate
//   - A tile (and B tile when pre-transposed) staged to LDS with
//     global_load_async_to_lds_b128 (ASYNCcnt), drained with s_wait_asynccnt
//   - LDS layouts are fragment-friendly: every lane's WMMA fragment is
//     contiguous 16B/32B chunks -> ds_load_b128, no scalar LDS loads
//   - all of M % 64, N % 64, K % 32 are 0 for every call in this pipeline
// Fragment layouts as loaded here (contiguous k; A and B share the order):
//   A 16x32 bf16 : lane L(0..15) row M=L : elems 0..15 = K0..15
//                  lane L+16     row M=L : elems 0..15 = K16..31
//   B 32x16 bf16 : lanes 0..15 N=lane, elem e = K e ; lanes 16..31 K = 16+e
//   C f32        : vgpr r, lanes 0..15 -> (M=r, N=lane); lanes 16..31 -> M=r+8
// ---------------------------------------------------------------------------
__global__ __launch_bounds__(256) void gemm_wmma_bf16(
    const bf16* __restrict__ A,  int lda, long aSB, long aSH,
    const bf16* __restrict__ Bm, int ldb, long bSB, long bSH, int bTrans,
    const float* __restrict__ bias, const float* __restrict__ resid,
    float* __restrict__ Cf, bf16* __restrict__ Cb, int ldc, long cSB, long cSH,
    int K, int batchH, float scale)
{
    const int z  = blockIdx.z;
    const int zb = z / batchH, zh = z % batchH;
    A  += zb * aSB + zh * aSH;
    Bm += zb * bSB + zh * bSH;
    const long cOff = zb * cSB + zh * cSH;

    const int tile_m = blockIdx.y * 64;
    const int tile_n = blockIdx.x * 64;
    const int tid  = threadIdx.x;
    const int lane = tid & 31;
    const int wid  = tid >> 5;
    const int wm   = wid >> 1;          // 0..3 : 16-row band
    const int wn   = wid & 1;           // 0..1 : 32-col band
    const int l16  = lane & 15;
    const int hv   = lane >> 4;         // lane half

    __shared__ __align__(16) bf16 As [64][40];   // [m][k], 64x32 + pad
    __shared__ __align__(16) bf16 BsT[64][40];   // [n][k], 64x32 + pad

    // staging coordinates: one 16B chunk per thread (64 rows x 32 k)
    const int sr = tid >> 2;            // 0..63
    const int sc = (tid & 3) * 8;       // 0,8,16,24
    const unsigned ldsA  = (unsigned)(size_t)&As [sr][sc];
    const unsigned ldsBT = (unsigned)(size_t)&BsT[sr][sc];
    // k-major B staging coordinates (transpose scatter, packed u32 stores)
    const int kb = (tid >> 4) * 2;      // 0,2,..,30
    const int nb = (tid & 15) * 4;      // 0,4,..,60

    v8f acc0 = {}; v8f acc1 = {};

    for (int k0 = 0; k0 < K; k0 += 32) {
        // ---- stage A tile (64x32) via async copy, no VGPR round trip
        {
            const bf16* g = A + (size_t)(tile_m + sr) * lda + k0 + sc;
            asm volatile("global_load_async_to_lds_b128 %0, %1, off"
                         :: "v"(ldsA), "v"(g) : "memory");
        }
        // ---- stage B tile (32x64) into n-major BsT
        if (bTrans) {
            // B stored N x K: rows map directly onto BsT -> async copy
            const bf16* g = Bm + (size_t)(tile_n + sr) * ldb + k0 + sc;
            asm volatile("global_load_async_to_lds_b128 %0, %1, off"
                         :: "v"(ldsBT), "v"(g) : "memory");
        } else {
            // B stored K x N: load two 8B rows, scatter 4 packed u32 stores
            const unsigned* g0 = (const unsigned*)(Bm + (size_t)(k0 + kb)     * ldb + tile_n + nb);
            const unsigned* g1 = (const unsigned*)(Bm + (size_t)(k0 + kb + 1) * ldb + tile_n + nb);
            unsigned q0x = g0[0], q0y = g0[1];
            unsigned q1x = g1[0], q1y = g1[1];
            unsigned* dst = (unsigned*)&BsT[nb][kb];   // row pitch = 20 u32
            dst[0]  = (q0x & 0xFFFFu) | (q1x << 16);
            dst[20] = (q0x >> 16)     | (q1x & 0xFFFF0000u);
            dst[40] = (q0y & 0xFFFFu) | (q1y << 16);
            dst[60] = (q0y >> 16)     | (q1y & 0xFFFF0000u);
        }
        // far prefetch of the next K tile (global_prefetch_b8)
        if (k0 + 32 < K) {
            __builtin_prefetch(A + (size_t)(tile_m + sr) * lda + k0 + 32, 0, 3);
        }
        asm volatile("s_wait_asynccnt 0x0" ::: "memory");
        __syncthreads();

        // ---- fragments: contiguous ds_load_b128 pairs
        v16bf af;
        {
            const bf16* ar = &As[wm * 16 + l16][0];
            v8bf a0 = *(const v8bf*)(ar + hv * 16);
            v8bf a1 = *(const v8bf*)(ar + hv * 16 + 8);
            af = __builtin_shufflevector(a0, a1, 0,1,2,3,4,5,6,7,8,9,10,11,12,13,14,15);
        }
        {
            const int nc = wn * 32 + l16;
            const bf16* b0p = &BsT[nc][hv * 16];
            v8bf b0a = *(const v8bf*)(b0p);
            v8bf b0b = *(const v8bf*)(b0p + 8);
            v16bf bq0 = __builtin_shufflevector(b0a, b0b, 0,1,2,3,4,5,6,7,8,9,10,11,12,13,14,15);
            const bf16* b1p = &BsT[nc + 16][hv * 16];
            v8bf b1a = *(const v8bf*)(b1p);
            v8bf b1b = *(const v8bf*)(b1p + 8);
            v16bf bq1 = __builtin_shufflevector(b1a, b1b, 0,1,2,3,4,5,6,7,8,9,10,11,12,13,14,15);
            acc0 = __builtin_amdgcn_wmma_f32_16x16x32_bf16(false, af, false, bq0,
                                                           (short)0, acc0, false, false);
            acc1 = __builtin_amdgcn_wmma_f32_16x16x32_bf16(false, af, false, bq1,
                                                           (short)0, acc1, false, false);
        }
        __syncthreads();
    }

    // ---- epilogue: scale + bias + residual, dual fp32/bf16 store
    const int gRowBase = tile_m + wm * 16 + hv * 8;
    const int gColBase = tile_n + wn * 32 + l16;
    #pragma unroll
    for (int r = 0; r < 8; ++r) {
        long row = gRowBase + r;
        #pragma unroll
        for (int nt = 0; nt < 2; ++nt) {
            long col = gColBase + nt * 16;
            float v = (nt ? acc1[r] : acc0[r]) * scale;
            if (bias)  v += bias[col];
            long o = cOff + row * (long)ldc + col;
            if (resid) v += resid[o];
            if (Cf) Cf[o] = v;
            if (Cb) Cb[o] = (bf16)v;
        }
    }
}

// ---------------------------------------------------------------------------
// Gated chunked scan (CHUNK=32). One thread per (b, channel), sequential in t.
// Exactly mirrors the reference step(): within-chunk cumulative product/sum
// with the 1e-6 / 1e-8 clips, carry = last element of each chunk.
// ---------------------------------------------------------------------------
__global__ void scan_kernel(const float* __restrict__ glogit,
                            const float* __restrict__ din,
                            bf16* __restrict__ states)
{
    int idx = blockIdx.x * blockDim.x + threadIdx.x;
    if (idx >= Bc * Dc) return;
    int b = idx / Dc, c = idx % Dc;
    const float* gp = glogit + (size_t)b * Tc * Dc + c;
    const float* dp = din    + (size_t)b * Tc * Dc + c;
    bf16*        sp = states + (size_t)b * Tc * Dc + c;
    float h = 0.f, cum_a = 1.f, cum_wb = 0.f;
    for (int t = 0; t < Tc; ++t) {
        size_t o = (size_t)t * Dc;
        float g  = 1.f / (1.f + __expf(-gp[o]));
        float dr = (1.f - g) * dp[o];
        cum_a  *= fmaxf(g, 1e-6f);
        cum_wb += dr / fmaxf(cum_a, 1e-8f);
        float out = cum_a * (h + cum_wb);
        sp[o] = (bf16)out;
        if ((t & 31) == 31) { h = out; cum_a = 1.f; cum_wb = 0.f; }
    }
}

// ---------------------------------------------------------------------------
// Row LayerNorm (N=768). One 256-thread block per row; fp32 + optional bf16.
// ---------------------------------------------------------------------------
__global__ __launch_bounds__(256) void ln_kernel(
    const float* __restrict__ in, const float* __restrict__ g,
    const float* __restrict__ bta, float* __restrict__ outf,
    bf16* __restrict__ outh, int N)
{
    long row = blockIdx.x;
    const float* r = in + row * N;
    float s = 0.f, s2 = 0.f;
    for (int c = threadIdx.x; c < N; c += blockDim.x) { float v = r[c]; s += v; s2 += v * v; }
    __shared__ float sh1[256], sh2[256];
    sh1[threadIdx.x] = s; sh2[threadIdx.x] = s2;
    __syncthreads();
    for (int o = 128; o > 0; o >>= 1) {
        if (threadIdx.x < o) { sh1[threadIdx.x] += sh1[threadIdx.x + o];
                               sh2[threadIdx.x] += sh2[threadIdx.x + o]; }
        __syncthreads();
    }
    float mean = sh1[0] / N;
    float var  = sh2[0] / N - mean * mean;
    float rstd = rsqrtf(var + 1e-5f);
    for (int c = threadIdx.x; c < N; c += blockDim.x) {
        float v = (r[c] - mean) * rstd * g[c] + bta[c];
        long o = row * N + c;
        if (outf) outf[o] = v;
        if (outh) outh[o] = (bf16)v;
    }
}

// ---------------------------------------------------------------------------
// Row softmax (one wave32 per row), fp32 in-place + bf16 shadow for PV GEMM.
// rows is always a multiple of 8 here.
// ---------------------------------------------------------------------------
__global__ __launch_bounds__(256) void softmax_rows_kernel(
    float* __restrict__ S, bf16* __restrict__ Sh, long rows, int L)
{
    long row = (long)blockIdx.x * 8 + (threadIdx.x >> 5);
    int lane = threadIdx.x & 31;
    float* p = S + row * (long)L;
    float mx = NEG_INF;
    for (int i = lane; i < L; i += 32) mx = fmaxf(mx, p[i]);
    for (int o = 16; o > 0; o >>= 1) mx = fmaxf(mx, __shfl_xor(mx, o));
    float sum = 0.f;
    for (int i = lane; i < L; i += 32) { float e = __expf(p[i] - mx); p[i] = e; sum += e; }
    for (int o = 16; o > 0; o >>= 1) sum += __shfl_xor(sum, o);
    float inv = 1.f / sum;
    for (int i = lane; i < L; i += 32) {
        float v = p[i] * inv;
        p[i] = v;
        Sh[row * (long)L + i] = (bf16)v;
    }
}

// ---------------------------------------------------------------------------
// Top-16 of a 512-wide row (one wave per row). Iterative wave-argmax with
// lowest-index tie-break (matches jax.lax.top_k ordering).
// ---------------------------------------------------------------------------
__global__ __launch_bounds__(256) void topk16_512_kernel(
    const float* __restrict__ S, float* __restrict__ val, int* __restrict__ idx)
{
    long row = (long)blockIdx.x * 8 + (threadIdx.x >> 5);
    int lane = threadIdx.x & 31;
    const float* p = S + row * (long)SUBK;
    float lv[16];
    #pragma unroll
    for (int i = 0; i < 16; ++i) lv[i] = p[lane * 16 + i];
    for (int k = 0; k < TOPK; ++k) {
        float bm = NEG_INF; int bi = 0;
        #pragma unroll
        for (int i = 0; i < 16; ++i) if (lv[i] > bm) { bm = lv[i]; bi = i; }
        float m = bm; int mi = lane * 16 + bi;
        for (int o = 16; o > 0; o >>= 1) {
            float om = __shfl_xor(m, o); int omi = __shfl_xor(mi, o);
            if (om > m || (om == m && omi < mi)) { m = om; mi = omi; }
        }
        if (lane == 0) { val[row * TOPK + k] = m; idx[row * TOPK + k] = mi; }
        if ((mi >> 4) == lane) lv[mi & 15] = NEG_INF;
    }
}

// ---------------------------------------------------------------------------
// PKM combine: 16x16 candidate scores -> top-16 -> softmax -> weighted gather
// from the fp32 value table -> bf16 mem row (VALD=256). One wave per row.
// ---------------------------------------------------------------------------
__global__ __launch_bounds__(256) void pkm_combine_kernel(
    const float* __restrict__ va, const int* __restrict__ ia,
    const float* __restrict__ vb, const int* __restrict__ ib,
    const float* __restrict__ values, bf16* __restrict__ mem)
{
    long row = (long)blockIdx.x * 8 + (threadIdx.x >> 5);
    int lane = threadIdx.x & 31;
    int wrow = threadIdx.x >> 5;
    const float* pva = va + row * TOPK; const int* pia = ia + row * TOPK;
    const float* pvb = vb + row * TOPK; const int* pib = ib + row * TOPK;

    float cs[8];
    #pragma unroll
    for (int k = 0; k < 8; ++k) {
        int c = lane * 8 + k;
        cs[k] = pva[c >> 4] + pvb[c & 15];
    }
    __shared__ float wsh[8][TOPK];
    __shared__ int   ish[8][TOPK];
    for (int k = 0; k < TOPK; ++k) {
        float bm = NEG_INF; int bi = 0;
        #pragma unroll
        for (int q = 0; q < 8; ++q) if (cs[q] > bm) { bm = cs[q]; bi = q; }
        float m = bm; int mi = lane * 8 + bi;
        for (int o = 16; o > 0; o >>= 1) {
            float om = __shfl_xor(m, o); int omi = __shfl_xor(mi, o);
            if (om > m || (om == m && omi < mi)) { m = om; mi = omi; }
        }
        if (lane == 0) {
            wsh[wrow][k] = m;
            ish[wrow][k] = pia[mi >> 4] * SUBK + pib[mi & 15];
        }
        if ((mi >> 3) == lane) cs[mi & 7] = NEG_INF;
    }
    __syncthreads();
    if (lane == 0) {                       // softmax over the 16 winners
        float mx = wsh[wrow][0];
        for (int k = 1; k < TOPK; ++k) mx = fmaxf(mx, wsh[wrow][k]);
        float ssum = 0.f;
        for (int k = 0; k < TOPK; ++k) { float e = __expf(wsh[wrow][k] - mx); wsh[wrow][k] = e; ssum += e; }
        float inv = 1.f / ssum;
        for (int k = 0; k < TOPK; ++k) wsh[wrow][k] *= inv;
    }
    __syncthreads();
    float acc[8] = {0, 0, 0, 0, 0, 0, 0, 0};
    for (int k = 0; k < TOPK; ++k) {
        float wgt = wsh[wrow][k];
        long base = (long)ish[wrow][k] * VALD;
        #pragma unroll
        for (int e = 0; e < 8; ++e) acc[e] += wgt * values[base + e * 32 + lane];
    }
    #pragma unroll
    for (int e = 0; e < 8; ++e) mem[row * VALD + e * 32 + lane] = (bf16)acc[e];
}

// ---------------------------------------------------------------------------
// Host launch
// ---------------------------------------------------------------------------
extern "C" void kernel_launch(void* const* d_in, const int* in_sizes, int n_in,
                              void* d_out, int out_size, void* d_ws, size_t ws_size,
                              hipStream_t stream)
{
    (void)in_sizes; (void)n_in; (void)out_size; (void)ws_size;

    const int*   tokens  = (const int*)d_in[0];
    const float* tables  = (const float*)d_in[1];
    const float* gate_w  = (const float*)d_in[2];
    const float* gate_b  = (const float*)d_in[3];
    const float* in_w    = (const float*)d_in[4];
    const float* in_b    = (const float*)d_in[5];
    const float* out_w   = (const float*)d_in[6];
    const float* out_b   = (const float*)d_in[7];
    const float* gs_ln_g = (const float*)d_in[8];
    const float* gs_ln_b = (const float*)d_in[9];
    const float* inducing= (const float*)d_in[10];
    const float* isab_w  = (const float*)d_in[11];
    const float* isab_b  = (const float*)d_in[12];
    const float* is_ln_g = (const float*)d_in[13];
    const float* is_ln_b = (const float*)d_in[14];
    const float* q_w     = (const float*)d_in[15];
    const float* q_b     = (const float*)d_in[16];
    const float* cb_a    = (const float*)d_in[17];
    const float* cb_b    = (const float*)d_in[18];
    const float* values  = (const float*)d_in[19];
    const float* po_w    = (const float*)d_in[20];
    const float* po_b    = (const float*)d_in[21];
    const float* pk_ln_g = (const float*)d_in[22];
    const float* pk_ln_b = (const float*)d_in[23];

    // ---- bump allocator over d_ws
    char* base = (char*)d_ws;
    size_t off = 0;
    auto alloc = [&](size_t bytes) -> void* {
        void* p = base + off;
        off = (off + bytes + 255) & ~(size_t)255;
        return p;
    };
    const size_t MD = (size_t)MT * Dc;

    float* xf   = (float*)alloc(MD * 4);          // hash embeddings fp32
    bf16*  xh   = (bf16*) alloc(MD * 2);
    float* x1f  = (float*)alloc(MD * 4);          // after gated-scan LN
    bf16*  x1h  = (bf16*) alloc(MD * 2);
    float* x2f  = (float*)alloc(MD * 4);          // after ISAB LN
    bf16*  x2h  = (bf16*) alloc(MD * 2);
    float* tf   = (float*)alloc(MD * 4);          // shared pre-LN temp
    float* gl   = (float*)alloc(MD * 4);          // gate logits; reused as scores_a
    float* di   = (float*)alloc(MD * 4);          // drive input; reused as scores_b
    bf16*  sth  = (bf16*) alloc(MD * 2);          // scan states (bf16)

    bf16* wgate_h = (bf16*)alloc((size_t)Dc * Dc * 2);
    bf16* win_h   = (bf16*)alloc((size_t)Dc * Dc * 2);
    bf16* wout_h  = (bf16*)alloc((size_t)Dc * Dc * 2);
    bf16* wis_h   = (bf16*)alloc((size_t)7 * Dc * Dc * 2);
    bf16* wq_h    = (bf16*)alloc((size_t)Dc * 2 * KEYD * 2);
    bf16* wcba_h  = (bf16*)alloc((size_t)SUBK * KEYD * 2);
    bf16* wcbb_h  = (bf16*)alloc((size_t)SUBK * KEYD * 2);
    bf16* wpo_h   = (bf16*)alloc((size_t)VALD * Dc * 2);
    bf16* wind_h  = (bf16*)alloc((size_t)PTS * Dc * 2);

    bf16* qI_h = (bf16*)alloc((size_t)PTS * Dc * 2);
    bf16* xk_h = (bf16*)alloc(MD * 2);
    bf16* xv_h = (bf16*)alloc(MD * 2);
    bf16* xq_h = (bf16*)alloc(MD * 2);

    const size_t S1N = (size_t)Bc * NHEAD * PTS * Tc;   // 12.58M
    float* S1f = (float*)alloc(S1N * 4);
    bf16*  S1h = (bf16*) alloc(S1N * 2);
    bf16*  Hs_h = (bf16*)alloc((size_t)Bc * PTS * Dc * 2);
    bf16*  Hk_h = (bf16*)alloc((size_t)Bc * PTS * Dc * 2);
    bf16*  Hv_h = (bf16*)alloc((size_t)Bc * PTS * Dc * 2);
    float* S2f = (float*)alloc(S1N * 4);
    bf16*  S2h = (bf16*) alloc(S1N * 2);
    bf16*  att_h = (bf16*)alloc(MD * 2);

    bf16*  qh  = (bf16*) alloc((size_t)MT * 2 * KEYD * 2);
    float* va  = (float*)alloc((size_t)MT * TOPK * 4);
    int*   ia  = (int*)  alloc((size_t)MT * TOPK * 4);
    float* vb  = (float*)alloc((size_t)MT * TOPK * 4);
    int*   ib  = (int*)  alloc((size_t)MT * TOPK * 4);
    bf16*  mem_h = (bf16*)alloc((size_t)MT * VALD * 2);

    // ---- helpers
    auto cvt = [&](const float* s, bf16* d, long n) {
        int blocks = (int)((n + 255) / 256);
        cvt_bf16_kernel<<<blocks, 256, 0, stream>>>(s, d, n);
    };
    auto gemm = [&](const bf16* A, int lda, long aSB, long aSH,
                    const bf16* Bm, int ldb, long bSB, long bSH, int bT,
                    const float* bias, const float* resid,
                    float* Cf, bf16* Cb, int ldc, long cSB, long cSH,
                    int M, int N, int K, int Z, int batchH, float scale) {
        dim3 g(N / 64, M / 64, Z);
        gemm_wmma_bf16<<<g, 256, 0, stream>>>(A, lda, aSB, aSH, Bm, ldb, bSB, bSH, bT,
                                              bias, resid, Cf, Cb, ldc, cSB, cSH,
                                              K, batchH, scale);
    };

    // ---- 0. weight bf16 shadows
    cvt(gate_w,  wgate_h, (long)Dc * Dc);
    cvt(in_w,    win_h,   (long)Dc * Dc);
    cvt(out_w,   wout_h,  (long)Dc * Dc);
    cvt(isab_w,  wis_h,   (long)7 * Dc * Dc);
    cvt(q_w,     wq_h,    (long)Dc * 2 * KEYD);
    cvt(cb_a,    wcba_h,  (long)SUBK * KEYD);
    cvt(cb_b,    wcbb_h,  (long)SUBK * KEYD);
    cvt(po_w,    wpo_h,   (long)VALD * Dc);
    cvt(inducing,wind_h,  (long)PTS * Dc);

    // ---- 1. hash embeddings
    hash_embed_kernel<<<MT, 256, 0, stream>>>(tokens, tables, xf, xh);

    // ---- 2. gated scan block
    gemm(xh, Dc,0,0, wgate_h, Dc,0,0,0, gate_b, nullptr, gl, nullptr, Dc,0,0,
         MT, Dc, Dc, 1, 1, 1.f);
    gemm(xh, Dc,0,0, win_h,   Dc,0,0,0, in_b,   nullptr, di, nullptr, Dc,0,0,
         MT, Dc, Dc, 1, 1, 1.f);
    scan_kernel<<<(Bc * Dc + 255) / 256, 256, 0, stream>>>(gl, di, sth);
    gemm(sth, Dc,0,0, wout_h, Dc,0,0,0, out_b, xf, tf, nullptr, Dc,0,0,
         MT, Dc, Dc, 1, 1, 1.f);
    ln_kernel<<<MT, 256, 0, stream>>>(tf, gs_ln_g, gs_ln_b, x1f, x1h, Dc);

    // ---- 3. ISAB
    const long WD = (long)Dc * Dc;
    // projections
    gemm(wind_h, Dc,0,0, wis_h + 0*WD, Dc,0,0,0, isab_b + 0*Dc, nullptr,
         nullptr, qI_h, Dc,0,0, PTS, Dc, Dc, 1, 1, 1.f);
    gemm(x1h, Dc,0,0, wis_h + 1*WD, Dc,0,0,0, isab_b + 1*Dc, nullptr,
         nullptr, xk_h, Dc,0,0, MT, Dc, Dc, 1, 1, 1.f);
    gemm(x1h, Dc,0,0, wis_h + 2*WD, Dc,0,0,0, isab_b + 2*Dc, nullptr,
         nullptr, xv_h, Dc,0,0, MT, Dc, Dc, 1, 1, 1.f);
    gemm(x1h, Dc,0,0, wis_h + 3*WD, Dc,0,0,0, isab_b + 3*Dc, nullptr,
         nullptr, xq_h, Dc,0,0, MT, Dc, Dc, 1, 1, 1.f);
    // mha1 scores: [b,h] (64 x 2048) = qI(64x64) @ xk(2048x64)^T, scale 1/8
    gemm(qI_h, Dc, 0, HD,
         xk_h, Dc, (long)Tc * Dc, HD, 1,
         nullptr, nullptr, S1f, nullptr, Tc, (long)NHEAD * PTS * Tc, (long)PTS * Tc,
         PTS, Tc, HD, Bc * NHEAD, NHEAD, 0.125f);
    softmax_rows_kernel<<<(Bc * NHEAD * PTS) / 8, 256, 0, stream>>>(
        S1f, S1h, (long)Bc * NHEAD * PTS, Tc);
    // mha1 PV: Hs[b,:, h*64..] (64x64) = A(64x2048) @ xv(2048x64)
    gemm(S1h, Tc, (long)NHEAD * PTS * Tc, (long)PTS * Tc,
         xv_h, Dc, (long)Tc * Dc, HD, 0,
         nullptr, nullptr, nullptr, Hs_h, Dc, (long)PTS * Dc, HD,
         PTS, HD, Tc, Bc * NHEAD, NHEAD, 1.f);
    // projections of Hs (B*PTS = 512 rows)
    gemm(Hs_h, Dc,0,0, wis_h + 4*WD, Dc,0,0,0, isab_b + 4*Dc, nullptr,
         nullptr, Hk_h, Dc,0,0, Bc * PTS, Dc, Dc, 1, 1, 1.f);
    gemm(Hs_h, Dc,0,0, wis_h + 5*WD, Dc,0,0,0, isab_b + 5*Dc, nullptr,
         nullptr, Hv_h, Dc,0,0, Bc * PTS, Dc, Dc, 1, 1, 1.f);
    // mha2 scores: [b,h] (2048 x 64) = xq(2048x64) @ Hk(64x64)^T
    gemm(xq_h, Dc, (long)Tc * Dc, HD,
         Hk_h, Dc, (long)PTS * Dc, HD, 1,
         nullptr, nullptr, S2f, nullptr, PTS, (long)NHEAD * Tc * PTS, (long)Tc * PTS,
         Tc, PTS, HD, Bc * NHEAD, NHEAD, 0.125f);
    softmax_rows_kernel<<<(Bc * NHEAD * Tc) / 8, 256, 0, stream>>>(
        S2f, S2h, (long)Bc * NHEAD * Tc, PTS);
    // mha2 PV: att[b,:, h*64..] (2048x64) = A(2048x64) @ Hv(64x64)
    gemm(S2h, PTS, (long)NHEAD * Tc * PTS, (long)Tc * PTS,
         Hv_h, Dc, (long)PTS * Dc, HD, 0,
         nullptr, nullptr, nullptr, att_h, Dc, (long)Tc * Dc, HD,
         Tc, HD, PTS, Bc * NHEAD, NHEAD, 1.f);
    // output projection + residual + LN
    gemm(att_h, Dc,0,0, wis_h + 6*WD, Dc,0,0,0, isab_b + 6*Dc, x1f,
         tf, nullptr, Dc,0,0, MT, Dc, Dc, 1, 1, 1.f);
    ln_kernel<<<MT, 256, 0, stream>>>(tf, is_ln_g, is_ln_b, x2f, x2h, Dc);

    // ---- 4. PKM
    gemm(x2h, Dc,0,0, wq_h, 2 * KEYD,0,0,0, q_b, nullptr,
         nullptr, qh, 2 * KEYD,0,0, MT, 2 * KEYD, Dc, 1, 1, 1.f);
    // scores_a = qa @ cb_a^T ; scores_b = qb @ cb_b^T   (reuse gl/di buffers)
    gemm(qh,          2 * KEYD,0,0, wcba_h, KEYD,0,0,1, nullptr, nullptr,
         gl, nullptr, SUBK,0,0, MT, SUBK, KEYD, 1, 1, 1.f);
    gemm(qh + KEYD,   2 * KEYD,0,0, wcbb_h, KEYD,0,0,1, nullptr, nullptr,
         di, nullptr, SUBK,0,0, MT, SUBK, KEYD, 1, 1, 1.f);
    topk16_512_kernel<<<MT / 8, 256, 0, stream>>>(gl, va, ia);
    topk16_512_kernel<<<MT / 8, 256, 0, stream>>>(di, vb, ib);
    pkm_combine_kernel<<<MT / 8, 256, 0, stream>>>(va, ia, vb, ib, values, mem_h);
    // output projection + residual + final LN -> d_out (fp32)
    gemm(mem_h, VALD,0,0, wpo_h, Dc,0,0,0, po_b, x2f,
         tf, nullptr, Dc,0,0, MT, Dc, VALD, 1, 1, 1.f);
    ln_kernel<<<MT, 256, 0, stream>>>(tf, pk_ln_g, pk_ln_b, (float*)d_out, nullptr, Dc);
}
